// CombinedGNN_85744727097865
// MI455X (gfx1250) — compile-verified, matrix operand order checked by reading, not run on hardware
//
#include <hip/hip_runtime.h>

// ---------------- WMMA types ----------------
typedef __attribute__((ext_vector_type(16))) __bf16 v16bf;
typedef __attribute__((ext_vector_type(8)))  float  v8f;

union BFrag { unsigned int u[8]; v16bf v; };

__device__ __forceinline__ unsigned short f2bf(float f) {
  unsigned u = __float_as_uint(f);
  unsigned r = u + 0x7FFFu + ((u >> 16) & 1u);   // round-to-nearest-even
  return (unsigned short)(r >> 16);
}
__device__ __forceinline__ unsigned int pack2bf(float a, float b) {
  return (unsigned int)f2bf(a) | ((unsigned int)f2bf(b) << 16);
}

// ordered-float encoding for atomicMax on negative values
__device__ __forceinline__ unsigned int encf(float f) {
  unsigned u = __float_as_uint(f);
  return (u & 0x80000000u) ? ~u : (u | 0x80000000u);
}
__device__ __forceinline__ float decf(unsigned int u) {
  unsigned v = (u & 0x80000000u) ? (u & 0x7FFFFFFFu) : ~u;
  return __uint_as_float(v);
}
__device__ __forceinline__ float leaky(float x) { return x > 0.f ? x : 0.2f * x; }

// ---------------- CDNA5 async global->LDS DMA (ASYNCcnt path) ----------------
// LDS destination address is taken from the VDST VGPR; low 32 bits of a generic shared pointer are the LDS offset.
__device__ __forceinline__ void async_copy_b128(unsigned lds_off, unsigned long long gaddr) {
  asm volatile("global_load_async_to_lds_b128 %0, %1, off"
               :: "v"(lds_off), "v"(gaddr) : "memory");
}
__device__ __forceinline__ void wait_asynccnt0() {
  asm volatile("s_wait_asynccnt 0x0" ::: "memory");
}

// ---------------- utility fills ----------------
__global__ void fill_f32(float* __restrict__ p, float v, long n) {
  long i = (long)blockIdx.x * blockDim.x + threadIdx.x;
  if (i < n) p[i] = v;
}
__global__ void fill_u32(unsigned int* __restrict__ p, unsigned int v, long n) {
  long i = (long)blockIdx.x * blockDim.x + threadIdx.x;
  if (i < n) p[i] = v;
}

// ---------------- weight prepack: W[K,NCOL] f32 -> Wp[n][kp] packed bf16 pairs ----------------
// Coalesced: consecutive threads read consecutive n within a K-row.
__global__ void prepack_wt(const float* __restrict__ W, unsigned int* __restrict__ Wp,
                           int K, int NCOL) {
  int idx = blockIdx.x * blockDim.x + threadIdx.x;
  int KP = K >> 1;
  if (idx >= NCOL * KP) return;
  int kp = idx / NCOL, n = idx - kp * NCOL;
  Wp[n * KP + kp] = pack2bf(W[(2 * kp) * NCOL + n], W[(2 * kp + 1) * NCOL + n]);
}

// ---------------- degree / norm ----------------
__global__ void deg_kernel(const int* __restrict__ dst, float* __restrict__ deg, int E) {
  int e = blockIdx.x * blockDim.x + threadIdx.x;
  if (e < E) atomicAdd(&deg[dst[e]], 1.0f);
}
__global__ void dinv_kernel(const float* __restrict__ deg, float* __restrict__ dinv, int n) {
  int i = blockIdx.x * blockDim.x + threadIdx.x;
  if (i < n) dinv[i] = rsqrtf(deg[i] + 1.0f);   // +1 self-loop
}

// ---------------- generic bf16 WMMA GEMM: Out[M,NCOL] = A[M,K] @ W[K,NCOL] ----------------
// blockDim = 128 (4 waves); each wave computes a 16-row x NCOL tile; block covers 64 rows.
// W arrives prepacked/transposed (bf16 pairs) and is staged via async LDS-DMA, overlapped
// with VALU f32->bf16 conversion of the A tile.
template<int K, int NCOL, bool ACCUM>
__global__ __launch_bounds__(128) void gemm_bf16(const float* __restrict__ A,
                                                 const unsigned int* __restrict__ Wp,
                                                 float* __restrict__ Out, int M) {
  constexpr int KP = K / 2;       // packed bf16 pairs along K
  constexpr int NT = NCOL / 16;   // 16-col WMMA tiles
  __shared__ __align__(16) unsigned int lds_a[64 * KP];     // [row][kp]
  __shared__ __align__(16) unsigned int lds_w[NCOL * KP];   // [n][kp]  (W transposed)
  const int tid = threadIdx.x;
  const int mbase = blockIdx.x * 64;

  // kick off async DMA of prepacked W^T (b128 per lane; count is a multiple of 128)
  {
    unsigned wbase = (unsigned)(size_t)(void*)lds_w;
    unsigned long long gbase = (unsigned long long)(size_t)Wp;
    for (int idx = tid; idx < (NCOL * KP) / 4; idx += 128)
      async_copy_b128(wbase + (unsigned)idx * 16u, gbase + (unsigned long long)idx * 16u);
  }
  // stage A tile as packed bf16 while the DMA is in flight
  for (int idx = tid; idx < 64 * KP; idx += 128) {
    int row = idx / KP, kp = idx % KP;
    int gr = mbase + row;
    unsigned int pv = 0u;
    if (gr < M) {
      const float2 f = *(const float2*)(A + (long)gr * K + 2 * kp);
      pv = pack2bf(f.x, f.y);
    }
    lds_a[idx] = pv;
  }
  wait_asynccnt0();
  __syncthreads();

  const int wave = tid >> 5;
  const int lane = tid & 31;
  const int hf = lane >> 4;     // lane half selects K-offset (+8 / +24)
  const int r = lane & 15;      // A: row within tile; B/C/D: column within tile
  const int rowbase = mbase + wave * 16;
  const unsigned int* arow = &lds_a[(wave * 16 + r) * KP];

  v8f acc[NT];
#pragma unroll
  for (int t = 0; t < NT; ++t) {
    if (ACCUM) {
#pragma unroll
      for (int j = 0; j < 8; ++j) {
        int row = rowbase + j + 8 * hf;
        acc[t][j] = (row < M) ? Out[(long)row * NCOL + t * 16 + r] : 0.f;
      }
    } else {
#pragma unroll
      for (int j = 0; j < 8; ++j) acc[t][j] = 0.f;
    }
  }

#pragma unroll
  for (int kt = 0; kt < K / 32; ++kt) {
    BFrag a;
#pragma unroll
    for (int p = 0; p < 8; ++p) {
      int kp = ((p < 4) ? p : p + 4) + hf * 4 + kt * 16;  // pairs (K,K+1); K=(i%8)+(i/8)*16+hf*8
      a.u[p] = arow[kp];
    }
#pragma unroll
    for (int t = 0; t < NT; ++t) {
      BFrag b;
      const unsigned int* wrow = &lds_w[(t * 16 + r) * KP];
#pragma unroll
      for (int p = 0; p < 8; ++p) {
        int kp = ((p < 4) ? p : p + 4) + hf * 4 + kt * 16;
        b.u[p] = wrow[kp];
      }
      acc[t] = __builtin_amdgcn_wmma_f32_16x16x32_bf16(false, a.v, false, b.v,
                                                       (short)0, acc[t], false, false);
    }
  }

#pragma unroll
  for (int t = 0; t < NT; ++t) {
#pragma unroll
    for (int j = 0; j < 8; ++j) {
      int row = rowbase + j + 8 * hf;
      if (row < M) Out[(long)row * NCOL + t * 16 + r] = acc[t][j];
    }
  }
}

// ---------------- GCN scatter + finalize ----------------
__global__ __launch_bounds__(64) void gcn_scatter(const float* __restrict__ hlin,
                                                  const int* __restrict__ src,
                                                  const int* __restrict__ dst,
                                                  const float* __restrict__ dinv,
                                                  float* __restrict__ hacc, int E) {
  int e = blockIdx.x, c = threadIdx.x;
  int s = src[e], d = dst[e];
  float norm = dinv[s] * dinv[d];
  atomicAdd(&hacc[(long)d * 64 + c], hlin[(long)s * 64 + c] * norm);
}
__global__ __launch_bounds__(64) void gcn_finalize(float* __restrict__ hacc,
                                                   const float* __restrict__ hlin,
                                                   const float* __restrict__ dinv,
                                                   const float* __restrict__ bgcn, int n) {
  int i = blockIdx.x, c = threadIdx.x;
  float di = dinv[i];
  float v = hacc[(long)i * 64 + c] + hlin[(long)i * 64 + c] * di * di + bgcn[c];
  hacc[(long)i * 64 + c] = fmaxf(v, 0.f);   // in place -> h
}

// ---------------- GAT ----------------
__global__ __launch_bounds__(128) void att_kernel(const float* __restrict__ glin,
                                                  const float* __restrict__ att_src,
                                                  const float* __restrict__ att_dst,
                                                  float* __restrict__ a_s,
                                                  float* __restrict__ a_d, int n) {
  int node = blockIdx.x;
  int hd = threadIdx.x >> 5, lane = threadIdx.x & 31;
  const float* gr = glin + (long)node * 256 + hd * 64;
  float s0 = gr[lane] * att_src[hd * 64 + lane] + gr[lane + 32] * att_src[hd * 64 + lane + 32];
  float d0 = gr[lane] * att_dst[hd * 64 + lane] + gr[lane + 32] * att_dst[hd * 64 + lane + 32];
  for (int m = 16; m >= 1; m >>= 1) { s0 += __shfl_xor(s0, m); d0 += __shfl_xor(d0, m); }
  if (lane == 0) { a_s[(long)node * 4 + hd] = s0; a_d[(long)node * 4 + hd] = d0; }
}

__global__ void gat_max(const int* __restrict__ src, const int* __restrict__ dst,
                        const float* __restrict__ a_s, const float* __restrict__ a_d,
                        unsigned int* __restrict__ menc, int E) {
  long t = (long)blockIdx.x * blockDim.x + threadIdx.x;
  if (t >= (long)E * 4) return;
  int e = (int)(t >> 2), hd = (int)(t & 3);
  int s = src[e], d = dst[e];
  float v = leaky(a_s[(long)s * 4 + hd] + a_d[(long)d * 4 + hd]);
  atomicMax(&menc[(long)d * 4 + hd], encf(v));
}

__global__ void gat_selfmax(const float* __restrict__ a_s, const float* __restrict__ a_d,
                            unsigned int* __restrict__ menc, int n4) {
  int t = blockIdx.x * blockDim.x + threadIdx.x;
  if (t >= n4) return;
  float v = leaky(a_s[t] + a_d[t]);               // self-loop contribution to segment max
  float mv = decf(menc[t]);
  ((float*)menc)[t] = fmaxf(mv, v);               // decode in place to float m
}

__global__ __launch_bounds__(256) void gat_scatter(const int* __restrict__ src,
                                                   const int* __restrict__ dst,
                                                   const float* __restrict__ a_s,
                                                   const float* __restrict__ a_d,
                                                   const float* __restrict__ m,
                                                   const float* __restrict__ glin,
                                                   float* __restrict__ gacc,
                                                   float* __restrict__ z, int E) {
  int e = blockIdx.x, c = threadIdx.x, hd = c >> 6;
  int s = src[e], d = dst[e];
  float v = leaky(a_s[(long)s * 4 + hd] + a_d[(long)d * 4 + hd]);
  float w = __expf(v - m[(long)d * 4 + hd]);
  atomicAdd(&gacc[(long)d * 256 + c], glin[(long)s * 256 + c] * w);
  if ((c & 63) == 0) atomicAdd(&z[(long)d * 4 + hd], w);
}

__global__ __launch_bounds__(256) void gat_finalize(const float* __restrict__ glin,
                                                    const float* __restrict__ a_s,
                                                    const float* __restrict__ a_d,
                                                    const float* __restrict__ m,
                                                    const float* __restrict__ z,
                                                    const float* __restrict__ bgat,
                                                    float* __restrict__ gacc, int n) {
  int i = blockIdx.x, c = threadIdx.x, hd = c >> 6;
  float v = leaky(a_s[(long)i * 4 + hd] + a_d[(long)i * 4 + hd]);
  float wn = __expf(v - m[(long)i * 4 + hd]);
  float zt = z[(long)i * 4 + hd] + wn + 1e-16f;
  float g = (gacc[(long)i * 256 + c] + glin[(long)i * 256 + c] * wn) / zt + bgat[c];
  gacc[(long)i * 256 + c] = fmaxf(g, 0.f);        // in place -> g
}

// ---------------- SAGE ----------------
__global__ __launch_bounds__(256) void sage_scatter(const int* __restrict__ src,
                                                    const int* __restrict__ dst,
                                                    const float* __restrict__ g,
                                                    float* __restrict__ nsum, int E) {
  int e = blockIdx.x, c = threadIdx.x;
  atomicAdd(&nsum[(long)dst[e] * 256 + c], g[(long)src[e] * 256 + c]);
}
__global__ void sage_mean(float* __restrict__ nsum, const float* __restrict__ deg, long n256) {
  long i = (long)blockIdx.x * blockDim.x + threadIdx.x;
  if (i < n256) nsum[i] /= fmaxf(deg[i >> 8], 1.0f);
}
__global__ __launch_bounds__(64) void sage_finalize(const float* __restrict__ sacc,
                                                    const float* __restrict__ bs,
                                                    float* __restrict__ s, int n) {
  int i = blockIdx.x, c = threadIdx.x;
  s[(long)i * 64 + c] = fmaxf(sacc[(long)i * 64 + c] + bs[c], 0.f);
}

// ---------------- fused edge MLP: sigmoid(relu([s_src||s_dst]@W1 + b1)@W2 + b2) ----------------
// blockDim 256 = 8 waves; each wave: 16 edges; WMMA 16x16x32 bf16, 4 K-steps x 4 N-tiles.
// W1^T (prepacked bf16) staged via async LDS-DMA, overlapped with gather of s rows.
__global__ __launch_bounds__(256) void edge_mlp(const float* __restrict__ sfeat,
                                                const int* __restrict__ src,
                                                const int* __restrict__ dst,
                                                const unsigned int* __restrict__ W1p,
                                                const float* __restrict__ b1,
                                                const float* __restrict__ W2,
                                                const float* __restrict__ b2,
                                                float* __restrict__ out, int E) {
  __shared__ __align__(16) unsigned int lds_w[64 * 64];    // W1^T: [n][kp], K=128
  __shared__ __align__(16) unsigned int lds_a[8][16 * 64]; // per-wave A tile: [row][kp]
  const int tid = threadIdx.x;

  {
    unsigned wbase = (unsigned)(size_t)(void*)lds_w;
    unsigned long long gbase = (unsigned long long)(size_t)W1p;
    for (int idx = tid; idx < (64 * 64) / 4; idx += 256)   // 1024 b128 ops, 4 per lane
      async_copy_b128(wbase + (unsigned)idx * 16u, gbase + (unsigned long long)idx * 16u);
  }

  const int wave = tid >> 5, lane = tid & 31;
  const int ebase = blockIdx.x * 128 + wave * 16;
#pragma unroll
  for (int i = 0; i < 32; ++i) {
    int idx = i * 32 + lane;              // 0..1023
    int row = idx >> 6, kp = idx & 63;
    int e = ebase + row;
    unsigned int pv = 0u;
    if (e < E) {
      int c0 = 2 * kp;
      const float* srow = (c0 < 64) ? (sfeat + (long)src[e] * 64 + c0)
                                    : (sfeat + (long)dst[e] * 64 + (c0 - 64));
      pv = pack2bf(srow[0], srow[1]);
    }
    lds_a[wave][idx] = pv;
  }
  wait_asynccnt0();
  __syncthreads();

  const int hf = lane >> 4, r = lane & 15;
  const unsigned int* arow = &lds_a[wave][r * 64];
  v8f acc[4];
#pragma unroll
  for (int t = 0; t < 4; ++t)
#pragma unroll
    for (int j = 0; j < 8; ++j) acc[t][j] = 0.f;

#pragma unroll
  for (int kt = 0; kt < 4; ++kt) {
    BFrag a;
#pragma unroll
    for (int p = 0; p < 8; ++p) a.u[p] = arow[((p < 4) ? p : p + 4) + hf * 4 + kt * 16];
#pragma unroll
    for (int t = 0; t < 4; ++t) {
      BFrag bfr;
      const unsigned int* wrow = &lds_w[(t * 16 + r) * 64];
#pragma unroll
      for (int p = 0; p < 8; ++p) bfr.u[p] = wrow[((p < 4) ? p : p + 4) + hf * 4 + kt * 16];
      acc[t] = __builtin_amdgcn_wmma_f32_16x16x32_bf16(false, a.v, false, bfr.v,
                                                       (short)0, acc[t], false, false);
    }
  }

  // epilogue: bias + relu + dot(W2) + cross-lane reduce + sigmoid
  float b1v[4], w2v[4];
#pragma unroll
  for (int t = 0; t < 4; ++t) { b1v[t] = b1[t * 16 + r]; w2v[t] = W2[t * 16 + r]; }
  float part[8];
#pragma unroll
  for (int j = 0; j < 8; ++j) {
    float p = 0.f;
#pragma unroll
    for (int t = 0; t < 4; ++t) {
      float hv = fmaxf(acc[t][j] + b1v[t], 0.f);
      p += hv * w2v[t];
    }
    part[j] = p;
  }
  for (int mk = 1; mk < 16; mk <<= 1) {
#pragma unroll
    for (int j = 0; j < 8; ++j) part[j] += __shfl_xor(part[j], mk);
  }
  if (r == 0) {
    float bb2 = b2[0];
#pragma unroll
    for (int j = 0; j < 8; ++j) {
      int e = ebase + j + 8 * hf;
      if (e < E) out[e] = 1.f / (1.f + __expf(-(part[j] + bb2)));
    }
  }
}

// ---------------- host launch ----------------
static inline int cdiv_i(long a, long b) { return (int)((a + b - 1) / b); }

extern "C" void kernel_launch(void* const* d_in, const int* in_sizes, int n_in,
                              void* d_out, int out_size, void* d_ws, size_t ws_size,
                              hipStream_t stream) {
  const float* x        = (const float*)d_in[0];
  const int*   ei       = (const int*)  d_in[1];
  const float* W_gcn    = (const float*)d_in[2];
  const float* b_gcn    = (const float*)d_in[3];
  const float* W_gat    = (const float*)d_in[4];
  const float* att_src  = (const float*)d_in[5];
  const float* att_dst  = (const float*)d_in[6];
  const float* b_gat    = (const float*)d_in[7];
  const float* W_sage_l = (const float*)d_in[8];
  const float* b_sage   = (const float*)d_in[9];
  const float* W_sage_r = (const float*)d_in[10];
  const float* W1       = (const float*)d_in[11];
  const float* b1       = (const float*)d_in[12];
  const float* W2       = (const float*)d_in[13];
  const float* b2       = (const float*)d_in[14];
  const int N = in_sizes[0] / 128;
  const int E = in_sizes[1] / 2;
  const int* src = ei;
  const int* dst = ei + E;

  float* base = (float*)d_ws;
  float* hlin = base; base += (long)N * 64;    // GCN lin; later reused as SAGE pre-activation
  float* hacc = base; base += (long)N * 64;    // GCN agg -> h
  float* deg  = base; base += N;               // in-degree (no self-loop)
  float* dinv = base; base += N;
  float* glin = base; base += (long)N * 256;   // GAT lin; later reused as nsum/mean
  float* gacc = base; base += (long)N * 256;   // GAT agg -> g
  float* asrc = base; base += (long)N * 4;
  float* adst = base; base += (long)N * 4;
  float* mbuf = base; base += (long)N * 4;     // encoded max, then decoded float max
  float* zbuf = base; base += (long)N * 4;
  float* sbuf = base; base += (long)N * 64;    // SAGE output
  // prepacked bf16 weights (transposed, pair-packed) for async LDS-DMA staging
  unsigned int* ub = (unsigned int*)base;
  unsigned int* wp_gcn = ub; ub += 64 * 64;    // K=128 -> KP=64,  NCOL=64
  unsigned int* wp_gat = ub; ub += 256 * 32;   // K=64  -> KP=32,  NCOL=256
  unsigned int* wp_sl  = ub; ub += 64 * 128;   // K=256 -> KP=128, NCOL=64
  unsigned int* wp_sr  = ub; ub += 64 * 128;
  unsigned int* wp_w1  = ub; ub += 64 * 64;    // K=128 -> KP=64,  NCOL=64
  float* pred = (float*)d_out;

  // init accumulators (workspace is not re-zeroed by harness)
  fill_f32<<<cdiv_i((long)N * 64, 256), 256, 0, stream>>>(hacc, 0.f, (long)N * 64);
  fill_f32<<<cdiv_i(N, 256), 256, 0, stream>>>(deg, 0.f, N);
  fill_f32<<<cdiv_i((long)N * 256, 256), 256, 0, stream>>>(gacc, 0.f, (long)N * 256);
  fill_f32<<<cdiv_i((long)N * 4, 256), 256, 0, stream>>>(zbuf, 0.f, (long)N * 4);
  fill_u32<<<cdiv_i((long)N * 4, 256), 256, 0, stream>>>((unsigned int*)mbuf, 0x007FFFFFu, (long)N * 4); // enc(-inf)

  // prepack all weights (tiny, coalesced)
  prepack_wt<<<cdiv_i(64 * 64, 256), 256, 0, stream>>>(W_gcn,    wp_gcn, 128, 64);
  prepack_wt<<<cdiv_i(256 * 32, 256), 256, 0, stream>>>(W_gat,   wp_gat, 64, 256);
  prepack_wt<<<cdiv_i(64 * 128, 256), 256, 0, stream>>>(W_sage_l, wp_sl, 256, 64);
  prepack_wt<<<cdiv_i(64 * 128, 256), 256, 0, stream>>>(W_sage_r, wp_sr, 256, 64);
  prepack_wt<<<cdiv_i(64 * 64, 256), 256, 0, stream>>>(W1,        wp_w1, 128, 64);

  // ---- GCN ----
  deg_kernel<<<cdiv_i(E, 256), 256, 0, stream>>>(dst, deg, E);
  dinv_kernel<<<cdiv_i(N, 256), 256, 0, stream>>>(deg, dinv, N);
  gemm_bf16<128, 64, false><<<cdiv_i(N, 64), 128, 0, stream>>>(x, wp_gcn, hlin, N);
  gcn_scatter<<<E, 64, 0, stream>>>(hlin, src, dst, dinv, hacc, E);
  gcn_finalize<<<N, 64, 0, stream>>>(hacc, hlin, dinv, b_gcn, N);          // hacc = h

  // ---- GAT ----
  gemm_bf16<64, 256, false><<<cdiv_i(N, 64), 128, 0, stream>>>(hacc, wp_gat, glin, N);
  att_kernel<<<N, 128, 0, stream>>>(glin, att_src, att_dst, asrc, adst, N);
  gat_max<<<cdiv_i((long)E * 4, 256), 256, 0, stream>>>(src, dst, asrc, adst, (unsigned int*)mbuf, E);
  gat_selfmax<<<cdiv_i((long)N * 4, 256), 256, 0, stream>>>(asrc, adst, (unsigned int*)mbuf, N * 4);
  gat_scatter<<<E, 256, 0, stream>>>(src, dst, asrc, adst, mbuf, glin, gacc, zbuf, E);
  gat_finalize<<<N, 256, 0, stream>>>(glin, asrc, adst, mbuf, zbuf, b_gat, gacc, N);  // gacc = g

  // ---- SAGE ----
  fill_f32<<<cdiv_i((long)N * 256, 256), 256, 0, stream>>>(glin, 0.f, (long)N * 256); // glin = nsum
  sage_scatter<<<E, 256, 0, stream>>>(src, dst, gacc, glin, E);
  sage_mean<<<cdiv_i((long)N * 256, 256), 256, 0, stream>>>(glin, deg, (long)N * 256);
  gemm_bf16<256, 64, false><<<cdiv_i(N, 64), 128, 0, stream>>>(glin, wp_sl, hlin, N);
  gemm_bf16<256, 64, true ><<<cdiv_i(N, 64), 128, 0, stream>>>(gacc, wp_sr, hlin, N);
  sage_finalize<<<N, 64, 0, stream>>>(hlin, b_sage, sbuf, N);

  // ---- fused edge MLP + sigmoid ----
  edge_mlp<<<cdiv_i(E, 128), 256, 0, stream>>>(sbuf, src, dst, wp_w1, b1, W2, b2, pred, E);
}
